// MultiHeadAttention_73031623901885
// MI455X (gfx1250) — matrix operand order repaired and checked by reading, not compiled, not run on hardware
//
#include <hip/hip_runtime.h>
#include <hip/hip_bf16.h>
#include <stdint.h>

// ---------------------------------------------------------------------------
// MI455X / gfx1250 multi-head attention forward.
// B=4, T=2048, D_MODEL=1024, H=16, DK=64. All matmuls via
// v_wmma_f32_16x16x32_bf16 (wave32, f32 accumulate).  Flash-attention K tiles
// staged by the Tensor Data Mover (tensor_load_to_lds, TENSORcnt), Q tiles by
// global_load_async_to_lds_b128 (ASYNCcnt).
// ---------------------------------------------------------------------------

typedef __bf16 bf16_t;
typedef __attribute__((ext_vector_type(16))) __bf16 v16bf;
typedef __attribute__((ext_vector_type(8)))  __bf16 v8bf;
typedef __attribute__((ext_vector_type(4)))  __bf16 v4bf;
typedef __attribute__((ext_vector_type(8)))  float  v8f;
typedef __attribute__((ext_vector_type(4)))  float  f32x4;
typedef __attribute__((ext_vector_type(4)))  unsigned int u32x4;
typedef __attribute__((ext_vector_type(8)))  unsigned int u32x8;

#define WMMA_BF16(A, B, C)                                                    \
  __builtin_amdgcn_wmma_f32_16x16x32_bf16(false, (A), false, (B), (short)0,   \
                                          (C), false, false)

// Async global->LDS copy, GV mode (16 bytes / lane).  Tracked by ASYNCcnt.
__device__ __forceinline__ void async_b128(uint32_t lds_off, const void* g) {
  asm volatile("global_load_async_to_lds_b128 %0, %1, off"
               :
               : "v"(lds_off), "v"(g)
               : "memory");
}
__device__ __forceinline__ void wait_async0() {
  asm volatile("s_wait_asynccnt 0x0" ::: "memory");
}

// Two aligned 16-byte LDS chunks -> one v16bf fragment register group.
__device__ __forceinline__ v16bf ld16(const bf16_t* p0, const bf16_t* p1) {
  v8bf lo = *(const v8bf*)p0;
  v8bf hi = *(const v8bf*)p1;
  return __builtin_shufflevector(lo, hi, 0, 1, 2, 3, 4, 5, 6, 7, 8, 9, 10, 11,
                                 12, 13, 14, 15);
}

// A-matrix 16x32 bf16 fragment (ISA 7.12.2): lane (hl = lane>>4) holds row m,
// elements 0..7 = K[8*hl .. 8*hl+7], elements 8..15 = K[16+8*hl .. 23+8*hl].
// LDS tile layout: row-major [m][k], row stride `ld` elements.
__device__ __forceinline__ v16bf frag_a(const bf16_t* s, int ld, int m,
                                        int kbase, int hl) {
  const bf16_t* r = s + m * ld + kbase + 8 * hl;
  return ld16(r, r + 16);
}

// B-matrix 32x16 bf16 fragment: lane holds column n = lane&15, elements 0..7 =
// K[8*hl .. 8*hl+7], 8..15 = K[16+8*hl ..].  LDS tile layout: [n][k] (k contiguous).
__device__ __forceinline__ v16bf frag_b(const bf16_t* s, int ld, int n,
                                        int kbase, int hl) {
  const bf16_t* r = s + n * ld + kbase + 8 * hl;
  return ld16(r, r + 16);
}

// ---------------------------------------------------------------------------
// Kernel 1: fused QKV projection.  Y = X @ W^T + b, output bf16 in
// (B, H, T, dk) layout.  blockIdx.z selects which of Q/K/V.
// Tile: 128(M) x 128(N) x 32(K); 8 waves of 32; wave tile 32x64 (2x4 frags).
// ---------------------------------------------------------------------------
enum { GBM = 128, GBN = 128, GBK = 32, GLD = GBK + 8 };

__global__ __launch_bounds__(256) void proj_qkv_kernel(
    const float* __restrict__ Qin, const float* __restrict__ Kin,
    const float* __restrict__ Vin, const float* __restrict__ Wq_,
    const float* __restrict__ bq_, const float* __restrict__ Wk_,
    const float* __restrict__ bk_, const float* __restrict__ Wv_,
    const float* __restrict__ bv_, bf16_t* __restrict__ qo,
    bf16_t* __restrict__ ko, bf16_t* __restrict__ vo) {
  __shared__ __align__(16) bf16_t As[GBM * GLD];
  __shared__ __align__(16) bf16_t Bs[GBN * GLD];

  const float* X;
  const float* W;
  const float* bias;
  bf16_t* out;
  if (blockIdx.z == 0) {
    X = Qin; W = Wq_; bias = bq_; out = qo;
  } else if (blockIdx.z == 1) {
    X = Kin; W = Wk_; bias = bk_; out = ko;
  } else {
    X = Vin; W = Wv_; bias = bv_; out = vo;
  }

  const int tid = threadIdx.x;
  const int lane = tid & 31, wv = tid >> 5;
  const int hl = (lane >> 4) & 1, l16 = lane & 15;
  const int m0 = blockIdx.y * GBM;
  const int n0 = blockIdx.x * GBN;
  const int wm = (wv & 3) * 32;   // wave row offset inside tile
  const int wn = (wv >> 2) * 64;  // wave col offset inside tile

  v8f acc[2][4];
#pragma unroll
  for (int i = 0; i < 2; ++i)
#pragma unroll
    for (int j = 0; j < 4; ++j) acc[i][j] = (v8f){};

  for (int k0 = 0; k0 < 1024; k0 += GBK) {
    // Stage A (X tile, f32 -> bf16) and B (W tile, [n][k] = W row-major).
#pragma unroll
    for (int p = 0; p < 4; ++p) {
      int idx = (p * 256 + tid) * 4;  // 0..4095
      int mm = idx >> 5, kk = idx & 31;
      f32x4 x = *(const f32x4*)(X + (size_t)(m0 + mm) * 1024 + k0 + kk);
      bf16_t* da = As + mm * GLD + kk;
      da[0] = (bf16_t)x[0]; da[1] = (bf16_t)x[1];
      da[2] = (bf16_t)x[2]; da[3] = (bf16_t)x[3];
      f32x4 w = *(const f32x4*)(W + (size_t)(n0 + mm) * 1024 + k0 + kk);
      bf16_t* db = Bs + mm * GLD + kk;
      db[0] = (bf16_t)w[0]; db[1] = (bf16_t)w[1];
      db[2] = (bf16_t)w[2]; db[3] = (bf16_t)w[3];
    }
    __syncthreads();

    v16bf a0 = frag_a(As, GLD, wm + l16, 0, hl);
    v16bf a1 = frag_a(As, GLD, wm + 16 + l16, 0, hl);
#pragma unroll
    for (int j = 0; j < 4; ++j) {
      v16bf bf = frag_b(Bs, GLD, wn + j * 16 + l16, 0, hl);
      acc[0][j] = WMMA_BF16(a0, bf, acc[0][j]);
      acc[1][j] = WMMA_BF16(a1, bf, acc[1][j]);
    }
    __syncthreads();
  }

  // Epilogue: bias, convert, scatter into (B, H, T, dk).
#pragma unroll
  for (int i = 0; i < 2; ++i)
#pragma unroll
    for (int j = 0; j < 4; ++j) {
      int n = n0 + wn + j * 16 + l16;
      float bn = bias[n];
      int hh = n >> 6, dk = n & 63;
#pragma unroll
      for (int r = 0; r < 8; ++r) {
        int m = m0 + wm + i * 16 + r + 8 * hl;
        int b = m >> 11, t = m & 2047;
        out[(((size_t)b * 16 + hh) * 2048 + t) * 64 + dk] =
            (bf16_t)(acc[i][j][r] + bn);
      }
    }
}

// ---------------------------------------------------------------------------
// Kernel 2: flash attention.  One workgroup = one (b, h, 128-query tile).
// 8 waves; wave owns 16 query rows.  Online softmax in f32.
// K tiles staged by TDM (with pad -> 72-element LDS rows), Q by async loads.
// ---------------------------------------------------------------------------
enum { FBQ = 128, FBKV = 128, FDK = 64, FLDK = 72, FLDV = 136, FLDP = 136 };

__global__ __launch_bounds__(256) void flash_attn_kernel(
    const bf16_t* __restrict__ q, const bf16_t* __restrict__ k,
    const bf16_t* __restrict__ v, const int* __restrict__ mask,
    bf16_t* __restrict__ att) {
  __shared__ __align__(16) bf16_t Qs[FBQ * FLDK];     // [t][dk]
  __shared__ __align__(16) bf16_t Ks[FBKV * FLDK];    // [kv][dk] (B of Q.K^T)
  __shared__ __align__(16) bf16_t Vs[FDK * FLDV];     // [dk][kv] (B of P.V)
  __shared__ __align__(16) bf16_t Ps[8 * 16 * FLDP];  // per-wave P scratch

  const int tid = threadIdx.x;
  const int lane = tid & 31, wv = tid >> 5;
  const int hl = (lane >> 4) & 1, l16 = lane & 15;
  const int bi = blockIdx.z, hh = blockIdx.y;
  const int q0 = blockIdx.x * FBQ;
  const size_t hoff = (((size_t)bi * 16 + hh) * 2048) * 64;
  const bf16_t* qh = q + hoff;
  const bf16_t* kh = k + hoff;
  const bf16_t* vh = v + hoff;
  const int* mrow = mask + bi * 2048;

  const int srow = tid >> 1, sh = tid & 1;  // staging: 2 threads per 64-el row

  {  // stage Q tile via async global->LDS (ASYNCcnt path)
    const bf16_t* g = qh + (size_t)(q0 + srow) * 64 + sh * 32;
    uint32_t l = (uint32_t)(uintptr_t)(Qs + srow * FLDK + sh * 32);
#pragma unroll
    for (int c = 0; c < 4; ++c) async_b128(l + c * 16, g + c * 8);
    wait_async0();
  }

  // TDM descriptor group 1 (constant): 2D tensor of bf16, line length 64,
  // tile 64 x 128, LDS padding 4 dwords after every 32 dwords (-> stride 72).
  u32x8 g1;
  g1[0] = 0x07110000u;   // data_size=1(2B), pad_enable, interval=32dw, amt=4dw
  g1[1] = 64u << 16;     // tensor_dim0 = 64
  g1[2] = 2048u << 16;   // tensor_dim0 hi | tensor_dim1 = 2048 (lo16)
  g1[3] = 64u << 16;     // tensor_dim1 hi | tile_dim0 = 64
  g1[4] = 128u;          // tile_dim1 = 128 | tile_dim2 = 0
  g1[5] = 64u;           // tensor_dim0_stride = 64 (lo32)
  g1[6] = 0u;            // stride hi | tensor_dim1_stride lo16
  g1[7] = 0u;            // tensor_dim1_stride hi

  __syncthreads();
  v16bf aq0 = frag_a(Qs, FLDK, wv * 16 + l16, 0, hl);
  v16bf aq1 = frag_a(Qs, FLDK, wv * 16 + l16, 32, hl);

  v8f o0 = (v8f){}, o1 = (v8f){}, o2 = (v8f){}, o3 = (v8f){};
  float mst[8], lst[8];
#pragma unroll
  for (int r = 0; r < 8; ++r) { mst[r] = -1e30f; lst[r] = 0.0f; }

  for (int kv0 = 0; kv0 < 2048; kv0 += FBKV) {
    __syncthreads();
    if (wv == 0) {  // stage K tile via Tensor Data Mover (TENSORcnt path)
      uint64_t ga = (uint64_t)(uintptr_t)(kh + (size_t)kv0 * 64);
      u32x4 g0;
      g0[0] = 1u;                          // count=1, user descriptor
      g0[1] = (uint32_t)(uintptr_t)Ks;     // lds_addr
      g0[2] = (uint32_t)ga;                // global_addr[31:0]
      g0[3] = ((uint32_t)(ga >> 32) & 0x01FFFFFFu) | (2u << 30);  // type=2
      asm volatile("tensor_load_to_lds %0, %1" : : "s"(g0), "s"(g1)
                   : "memory");
    }
    {  // stage V tile transposed -> Vs[dk][kv]
      const v8bf* g = (const v8bf*)(vh + (size_t)(kv0 + srow) * 64 + sh * 32);
#pragma unroll
      for (int c4 = 0; c4 < 4; ++c4) {
        v8bf cc = g[c4];
#pragma unroll
        for (int e = 0; e < 8; ++e)
          Vs[(sh * 32 + c4 * 8 + e) * FLDV + srow] = cc[e];
      }
    }
    if (kv0 + FBKV < 2048) {  // prefetch next V tile -> global_prefetch_b8
      __builtin_prefetch(vh + (size_t)(kv0 + FBKV + srow) * 64, 0, 1);
    }
    if (wv == 0) __builtin_amdgcn_s_wait_tensorcnt(0);
    __syncthreads();

    // S = (Q tile) @ K^T : 8 column fragments of 16, two K-steps over dk.
    float s[8][8];
    float km[8];
#pragma unroll
    for (int j = 0; j < 8; ++j) {
      v16bf b0 = frag_b(Ks, FLDK, j * 16 + l16, 0, hl);
      v16bf b1 = frag_b(Ks, FLDK, j * 16 + l16, 32, hl);
      v8f c = (v8f){};
      c = WMMA_BF16(aq0, b0, c);
      c = WMMA_BF16(aq1, b1, c);
      km[j] = (mrow[kv0 + j * 16 + l16] != 0) ? 1.0f : 0.0f;
#pragma unroll
      for (int r = 0; r < 8; ++r)
        s[j][r] = (km[j] != 0.0f) ? c[r] * 0.125f : -1e30f;
    }

    // Row max across the 128-wide tile (frags, then 16-lane halves).
    float mx[8];
#pragma unroll
    for (int r = 0; r < 8; ++r) {
      mx[r] = s[0][r];
#pragma unroll
      for (int j = 1; j < 8; ++j) mx[r] = fmaxf(mx[r], s[j][r]);
    }
#pragma unroll
    for (int off = 8; off >= 1; off >>= 1)
#pragma unroll
      for (int r = 0; r < 8; ++r)
        mx[r] = fmaxf(mx[r], __shfl_xor(mx[r], off, 16));

    float mnew[8], corr[8], sm[8];
#pragma unroll
    for (int r = 0; r < 8; ++r) {
      mnew[r] = fmaxf(mst[r], mx[r]);
      corr[r] = __expf(mst[r] - mnew[r]);
      mst[r] = mnew[r];
      sm[r] = 0.0f;
    }

    // P = exp(S - m); masked columns forced to exactly 0.  Write bf16 P into
    // per-wave LDS to convert C-frag layout -> A-frag layout.
    bf16_t* Pw = Ps + wv * 16 * FLDP;
#pragma unroll
    for (int j = 0; j < 8; ++j)
#pragma unroll
      for (int r = 0; r < 8; ++r) {
        float p = km[j] * __expf(s[j][r] - mnew[r]);
        sm[r] += p;
        Pw[(r + 8 * hl) * FLDP + j * 16 + l16] = (bf16_t)p;
      }
#pragma unroll
    for (int off = 8; off >= 1; off >>= 1)
#pragma unroll
      for (int r = 0; r < 8; ++r) sm[r] += __shfl_xor(sm[r], off, 16);
#pragma unroll
    for (int r = 0; r < 8; ++r) lst[r] = lst[r] * corr[r] + sm[r];
#pragma unroll
    for (int r = 0; r < 8; ++r) {
      o0[r] *= corr[r]; o1[r] *= corr[r];
      o2[r] *= corr[r]; o3[r] *= corr[r];
    }

    // O += P @ V : 4 K-steps over the 128 kv positions, 4 dk-column frags.
#pragma unroll
    for (int st = 0; st < 4; ++st) {
      v16bf ap = frag_a(Pw, FLDP, l16, st * 32, hl);
      v16bf b0 = frag_b(Vs, FLDV, 0 + l16, st * 32, hl);
      v16bf b1 = frag_b(Vs, FLDV, 16 + l16, st * 32, hl);
      v16bf b2 = frag_b(Vs, FLDV, 32 + l16, st * 32, hl);
      v16bf b3 = frag_b(Vs, FLDV, 48 + l16, st * 32, hl);
      o0 = WMMA_BF16(ap, b0, o0);
      o1 = WMMA_BF16(ap, b1, o1);
      o2 = WMMA_BF16(ap, b2, o2);
      o3 = WMMA_BF16(ap, b3, o3);
    }
  }

  // Normalize and write combined-heads layout (B, T, H*dk) as bf16.
  float inv[8];
#pragma unroll
  for (int r = 0; r < 8; ++r) inv[r] = 1.0f / lst[r];
#pragma unroll
  for (int r = 0; r < 8; ++r) {
    int t = q0 + wv * 16 + r + 8 * hl;
    bf16_t* dst = att + ((size_t)bi * 2048 + t) * 1024 + hh * 64;
    dst[0 + l16]  = (bf16_t)(o0[r] * inv[r]);
    dst[16 + l16] = (bf16_t)(o1[r] * inv[r]);
    dst[32 + l16] = (bf16_t)(o2[r] * inv[r]);
    dst[48 + l16] = (bf16_t)(o3[r] * inv[r]);
  }
}

// ---------------------------------------------------------------------------
// Kernel 3: output projection.  out = att(bf16) @ Wo^T + bo, f32 output.
// ---------------------------------------------------------------------------
__global__ __launch_bounds__(256) void proj_out_kernel(
    const bf16_t* __restrict__ Ain, const float* __restrict__ W,
    const float* __restrict__ bias, float* __restrict__ Y) {
  __shared__ __align__(16) bf16_t As[GBM * GLD];
  __shared__ __align__(16) bf16_t Bs[GBN * GLD];

  const int tid = threadIdx.x;
  const int lane = tid & 31, wv = tid >> 5;
  const int hl = (lane >> 4) & 1, l16 = lane & 15;
  const int m0 = blockIdx.y * GBM;
  const int n0 = blockIdx.x * GBN;
  const int wm = (wv & 3) * 32;
  const int wn = (wv >> 2) * 64;

  v8f acc[2][4];
#pragma unroll
  for (int i = 0; i < 2; ++i)
#pragma unroll
    for (int j = 0; j < 4; ++j) acc[i][j] = (v8f){};

  for (int k0 = 0; k0 < 1024; k0 += GBK) {
#pragma unroll
    for (int p = 0; p < 4; ++p) {
      int idx = (p * 256 + tid) * 4;
      int mm = idx >> 5, kk = idx & 31;
      v4bf x = *(const v4bf*)(Ain + (size_t)(m0 + mm) * 1024 + k0 + kk);
      *(v4bf*)(As + mm * GLD + kk) = x;
      f32x4 w = *(const f32x4*)(W + (size_t)(n0 + mm) * 1024 + k0 + kk);
      bf16_t* db = Bs + mm * GLD + kk;
      db[0] = (bf16_t)w[0]; db[1] = (bf16_t)w[1];
      db[2] = (bf16_t)w[2]; db[3] = (bf16_t)w[3];
    }
    __syncthreads();

    v16bf a0 = frag_a(As, GLD, wm + l16, 0, hl);
    v16bf a1 = frag_a(As, GLD, wm + 16 + l16, 0, hl);
#pragma unroll
    for (int j = 0; j < 4; ++j) {
      v16bf bf = frag_b(Bs, GLD, wn + j * 16 + l16, 0, hl);
      acc[0][j] = WMMA_BF16(a0, bf, acc[0][j]);
      acc[1][j] = WMMA_BF16(a1, bf, acc[1][j]);
    }
    __syncthreads();
  }

#pragma unroll
  for (int i = 0; i < 2; ++i)
#pragma unroll
    for (int j = 0; j < 4; ++j) {
      int n = n0 + wn + j * 16 + l16;
      float bn = bias[n];
#pragma unroll
      for (int r = 0; r < 8; ++r) {
        int m = m0 + wm + i * 16 + r + 8 * hl;
        Y[(size_t)m * 1024 + n] = acc[i][j][r] + bn;
      }
    }
}

// ---------------------------------------------------------------------------
// Launcher.  Workspace: qbf | kbf | vbf (each (B,H,T,dk) bf16) | att
// ((B,T,D) bf16) = 64 MB total.
// ---------------------------------------------------------------------------
extern "C" void kernel_launch(void* const* d_in, const int* in_sizes, int n_in,
                              void* d_out, int out_size, void* d_ws,
                              size_t ws_size, hipStream_t stream) {
  const float* Q = (const float*)d_in[0];
  const float* K = (const float*)d_in[1];
  const float* V = (const float*)d_in[2];
  const int* mask = (const int*)d_in[3];
  const float* Wq = (const float*)d_in[4];
  const float* bq = (const float*)d_in[5];
  const float* Wk = (const float*)d_in[6];
  const float* bk = (const float*)d_in[7];
  const float* Wv = (const float*)d_in[8];
  const float* bv = (const float*)d_in[9];
  const float* Wo = (const float*)d_in[10];
  const float* bo = (const float*)d_in[11];
  float* out = (float*)d_out;

  const size_t ELEMS = (size_t)8192 * 1024;
  bf16_t* qbf = (bf16_t*)d_ws;
  bf16_t* kbf = qbf + ELEMS;
  bf16_t* vbf = kbf + ELEMS;
  bf16_t* att = vbf + ELEMS;

  dim3 blk(256, 1, 1);
  proj_qkv_kernel<<<dim3(8, 64, 3), blk, 0, stream>>>(
      Q, K, V, Wq, bq, Wk, bk, Wv, bv, qbf, kbf, vbf);
  flash_attn_kernel<<<dim3(16, 16, 4), blk, 0, stream>>>(qbf, kbf, vbf, mask,
                                                         att);
  proj_out_kernel<<<dim3(8, 64, 1), blk, 0, stream>>>(att, Wo, bo, out);
}
